// LNLinearActivationModel_59356448031118
// MI455X (gfx1250) — matrix operand order repaired and checked by reading, not compiled, not run on hardware
//
#include <hip/hip_runtime.h>
#include <hip/hip_bf16.h>

// ---------------------------------------------------------------------------
// LN + Linear(no bias) + sigmoid, MI455X (gfx1250, wave32, WMMA).
//   x  : [8192, 4096]  f32
//   W  : [16384, 4096] f32   (nn.Linear layout: [out, in])
//   out: [8192, 16384] f32
// Phase 1: row LayerNorm -> bf16 xn in d_ws (64 MB, fully L2-resident).
// Phase 2: double-buffered WMMA bf16 GEMM:
//   - A tile staged with GLOBAL_LOAD_ASYNC_TO_LDS_B128 (ASYNCcnt, no VGPRs)
//   - B tile fp32 -> bf16 via VGPRs (v_cvt_pk_bf16_f32) -> ds_store
//   - 8x v_wmma_f32_16x16x32_bf16 per wave per k-step, f32 accum
//   - fused sigmoid epilogue, nontemporal (TH=NT) output stores
// ---------------------------------------------------------------------------

typedef __attribute__((ext_vector_type(4)))  float        v4f;
typedef __attribute__((ext_vector_type(8)))  float        v8f;
typedef __attribute__((ext_vector_type(4)))  __bf16       v4bf;
typedef __attribute__((ext_vector_type(8)))  __bf16       v8bf;
typedef __attribute__((ext_vector_type(16))) __bf16       v16bf;

static constexpr int TOKENS = 8192;
static constexpr int KDIM   = 4096;
static constexpr int NDIM   = 16384;
static constexpr float LN_EPS = 1e-5f;

static __device__ __forceinline__ v16bf cat8(v8bf lo, v8bf hi) {
  return __builtin_shufflevector(lo, hi, 0, 1, 2, 3, 4, 5, 6, 7,
                                          8, 9, 10, 11, 12, 13, 14, 15);
}

// Async 16B copy global -> LDS (CDNA5, ASYNCcnt-tracked, GV addressing).
// Flat LDS addresses map to LDS by their low 32 bits (ISA 10.2 aperture rules).
static __device__ __forceinline__ void async_cp16(void* lds_ptr,
                                                  const void* gptr) {
  const unsigned lds_addr = (unsigned)(uintptr_t)lds_ptr;
  const unsigned long long gaddr = (unsigned long long)(uintptr_t)gptr;
  asm volatile("global_load_async_to_lds_b128 %0, %1, off"
               :
               : "v"(lds_addr), "v"(gaddr)
               : "memory");
}
static __device__ __forceinline__ void wait_async0() {
  asm volatile("s_wait_asynccnt 0x0" ::: "memory");
}

// ---------------------------------------------------------------------------
// Phase 1: LayerNorm, one row per workgroup (256 thr = 8 wave32), emit bf16.
// x is read-once -> nontemporal loads so it does not evict W/xn from L2.
// ---------------------------------------------------------------------------
__global__ __launch_bounds__(256) void ln_bf16_kernel(
    const float* __restrict__ x, __bf16* __restrict__ xn) {
  const size_t base = (size_t)blockIdx.x * KDIM;
  const float* xr = x + base;
  const int t = threadIdx.x;

  v4f v[4];
  float s = 0.f, ss = 0.f;
#pragma unroll
  for (int i = 0; i < 4; ++i) {
    v[i] = __builtin_nontemporal_load((const v4f*)(xr + i * 1024 + t * 4));
#pragma unroll
    for (int e = 0; e < 4; ++e) { s += v[i][e]; ss += v[i][e] * v[i][e]; }
  }

  // wave32 butterfly reduce
#pragma unroll
  for (int off = 16; off > 0; off >>= 1) {
    s  += __shfl_down(s,  off);
    ss += __shfl_down(ss, off);
  }
  __shared__ float red[2][8];
  const int wid = t >> 5, lane = t & 31;
  if (lane == 0) { red[0][wid] = s; red[1][wid] = ss; }
  __syncthreads();
  float st = 0.f, sst = 0.f;
#pragma unroll
  for (int w = 0; w < 8; ++w) { st += red[0][w]; sst += red[1][w]; }

  const float mean = st * (1.0f / KDIM);
  const float var  = sst * (1.0f / KDIM) - mean * mean;   // biased, torch LN
  const float rstd = rsqrtf(var + LN_EPS);

  __bf16* orow = xn + base;
#pragma unroll
  for (int i = 0; i < 4; ++i) {
    v4bf o;
#pragma unroll
    for (int e = 0; e < 4; ++e) o[e] = (__bf16)((v[i][e] - mean) * rstd);
    *(v4bf*)(orow + i * 1024 + t * 4) = o;   // 8-byte store; stays cached
  }
}

// ---------------------------------------------------------------------------
// Phase 2: WMMA GEMM, tile 128x128x32, 256 thr = 8 waves in 2(M)x4(N),
// double-buffered LDS, async A staging, fused sigmoid.
// Grid rasterized M-fast inside an N-strip: the 2 MB fp32 W strip is
// HBM-read once and L2-served for the other 63 M-tiles.
// ---------------------------------------------------------------------------
static constexpr int BM = 128, BN = 128, BK = 32;
static constexpr int LDSPAD = 8;              // bf16 elems; row stride 80 B
static constexpr int KTILES = KDIM / BK;      // 128

__global__ __launch_bounds__(256) void gemm_wmma_sigmoid_kernel(
    const __bf16* __restrict__ A,   // xn [TOKENS, KDIM] bf16 row-major
    const float*  __restrict__ W,   // [NDIM, KDIM] f32 row-major
    float* __restrict__ out) {      // [TOKENS, NDIM] f32
  __shared__ __bf16 As[2][BM][BK + LDSPAD];   // 2 x 10 KB
  __shared__ __bf16 Bs[2][BN][BK + LDSPAD];   // 2 x 10 KB

  const int tile = blockIdx.x;
  const int tm = tile & 63;          // M fast: 8192/128 = 64 M-tiles
  const int tn = tile >> 6;          // 16384/128 = 128 N-strips
  const int m0 = tm * BM, n0 = tn * BN;

  const int tid  = threadIdx.x;
  const int lane = tid & 31;
  const int wid  = tid >> 5;
  const int wm   = (wid & 1) * 64;   // wave row offset in tile
  const int wn   = (wid >> 1) * 32;  // wave col offset in tile
  const int ar   = lane & 15;        // fragment row/col within 16
  const int asel = lane >> 4;        // lane-half selector

  // --- per-thread staging coordinates ---
  // A tile: 128x32 bf16 = 512 x 16B chunks; thread handles chunks tid, tid+256
  const int a_r0 = tid >> 2,          a_c0 = (tid & 3) << 3;
  const int a_r1 = (tid + 256) >> 2,  a_c1 = ((tid + 256) & 3) << 3;
  // B tile: 128x32 f32 = 1024 x 16B chunks; thread handles tid + i*256
  int b_r[4], b_c[4];
#pragma unroll
  for (int i = 0; i < 4; ++i) {
    const int c = tid + i * 256;
    b_r[i] = c >> 3;
    b_c[i] = (c & 7) << 2;
  }

  v8f acc[4][2] = {};                // 64 VGPRs of f32 accum per lane

  // ---- prologue: stage tile 0 into buffer 0 ----
  {
    async_cp16(&As[0][a_r0][a_c0], A + (size_t)(m0 + a_r0) * KDIM + a_c0);
    async_cp16(&As[0][a_r1][a_c1], A + (size_t)(m0 + a_r1) * KDIM + a_c1);
#pragma unroll
    for (int i = 0; i < 4; ++i) {
      const v4f f = *(const v4f*)(W + (size_t)(n0 + b_r[i]) * KDIM + b_c[i]);
      v4bf h;
#pragma unroll
      for (int e = 0; e < 4; ++e) h[e] = (__bf16)f[e];
      *(v4bf*)(&Bs[0][b_r[i]][b_c[i]]) = h;
    }
    wait_async0();
  }
  __syncthreads();

  for (int kt = 0; kt < KTILES; ++kt) {
    const int buf = kt & 1, nb = buf ^ 1;
    const int kn = (kt + 1) * BK;
    const bool more = (kt + 1) < KTILES;

    // ---- issue next tile's loads before compute ----
    v4f breg[4];
    if (more) {
      async_cp16(&As[nb][a_r0][a_c0], A + (size_t)(m0 + a_r0) * KDIM + kn + a_c0);
      async_cp16(&As[nb][a_r1][a_c1], A + (size_t)(m0 + a_r1) * KDIM + kn + a_c1);
#pragma unroll
      for (int i = 0; i < 4; ++i) {
        const float* gp = W + (size_t)(n0 + b_r[i]) * KDIM + kn + b_c[i];
        breg[i] = *(const v4f*)gp;
        if (kn + BK < KDIM) __builtin_prefetch(gp + BK, 0, 1);
      }
    }

    // ---- compute on current buffer (ISA 7.12.2 fragment layouts) ----
    v16bf af[4];
#pragma unroll
    for (int i = 0; i < 4; ++i) {
      // A 16x32 bf16: lanes 0-15 row M=lane, K=0-7 & 16-23;
      //               lanes 16-31 same rows, K=8-15 & 24-31.
      const __bf16* p = &As[buf][wm + i * 16 + ar][asel * 8];
      af[i] = cat8(*(const v8bf*)p, *(const v8bf*)(p + 16));
    }
    v16bf bfrag[2];
#pragma unroll
    for (int j = 0; j < 2; ++j) {
      // B 32x16 bf16: lanes 0-15 col N=lane, K=0-7 & 16-23; lanes 16-31 K=8-15 & 24-31.
      const __bf16* p = &Bs[buf][wn + j * 16 + ar][asel * 8];
      bfrag[j] = cat8(*(const v8bf*)p, *(const v8bf*)(p + 16));
    }
#pragma unroll
    for (int i = 0; i < 4; ++i)
#pragma unroll
      for (int j = 0; j < 2; ++j)
        acc[i][j] = __builtin_amdgcn_wmma_f32_16x16x32_bf16(
            false, af[i], false, bfrag[j], (short)0, acc[i][j], false, false);

    // ---- finish staging next tile (loadcnt wait lands after the WMMAs) ----
    if (more) {
#pragma unroll
      for (int i = 0; i < 4; ++i) {
        v4bf h;
#pragma unroll
        for (int e = 0; e < 4; ++e) h[e] = (__bf16)breg[i][e];
        *(v4bf*)(&Bs[nb][b_r[i]][b_c[i]]) = h;
      }
      wait_async0();
    }
    __syncthreads();
  }

  // ---- epilogue: sigmoid, nontemporal f32 stores (write-once stream) ----
  // C/D layout: lanes 0-15 N=lane, VGPR r -> M=r; lanes 16-31 -> M=8+r.
  const int ncol  = n0 + wn + ar;
  const int mbase = m0 + wm + asel * 8;
#pragma unroll
  for (int i = 0; i < 4; ++i) {
#pragma unroll
    for (int j = 0; j < 2; ++j) {
      const int n = ncol + j * 16;
#pragma unroll
      for (int r = 0; r < 8; ++r) {
        const int m = mbase + i * 16 + r;
        const float y = acc[i][j][r];
        __builtin_nontemporal_store(1.0f / (1.0f + __expf(-y)),
                                    &out[(size_t)m * NDIM + n]);
      }
    }
  }
}

// ---------------------------------------------------------------------------
extern "C" void kernel_launch(void* const* d_in, const int* in_sizes, int n_in,
                              void* d_out, int out_size, void* d_ws,
                              size_t ws_size, hipStream_t stream) {
  const float* x = (const float*)d_in[0];   // [8192, 4096] f32
  const float* W = (const float*)d_in[1];   // [16384, 4096] f32
  float* out = (float*)d_out;               // [8192, 16384] f32
  __bf16* xn = (__bf16*)d_ws;               // 8192*4096*2 = 64 MB scratch

  ln_bf16_kernel<<<TOKENS, 256, 0, stream>>>(x, xn);

  const int grid = (TOKENS / BM) * (NDIM / BN);   // 64 * 128 = 8192
  gemm_wmma_sigmoid_kernel<<<grid, 256, 0, stream>>>(xn, W, out);
}
